// VQWorldModel_47828755808927
// MI455X (gfx1250) — matrix operand order repaired and checked by reading, not compiled, not run on hardware
//
#include <hip/hip_runtime.h>
#include <hip/hip_bf16.h>
#include <math.h>

// ---------------------------------------------------------------------------
// VQ-VAE world model forward on gfx1250 (CDNA5, wave32, WMMA 16x16x32 f16)
//
// All dense math runs through a 4-wave WMMA GEMM pattern:
//   - A tile (16 x 32 fp32) staged cooperatively to LDS with aligned float4
//     loads, converted to fp16 in fragment order, re-read as one v16h/lane.
//   - B pre-packed to WMMA fragment order: one aligned v16h load per k-step.
//   - Each wave owns 4 N-tiles; the WMMA loop is UNCONDITIONAL (tile index
//     clamped for the B address) so EXEC stays all-ones, accumulators stay
//     pinned in registers, and no saveexec/phi-copy storms appear.
// ---------------------------------------------------------------------------

typedef __attribute__((ext_vector_type(16))) _Float16 v16h;
typedef __attribute__((ext_vector_type(8)))  float    v8f;
typedef __attribute__((ext_vector_type(4)))  float    v4f;

// WMMA fragment maps used here (wave32; A/B k order is this code's contiguous choice, not the ISA table):
// A (16x32 f16): row M = lane%16; half e -> K = (lane>=16)*16 + e
// B (32x16 f16): col N = lane%16; half e -> K = (lane>=16)*16 + e
// C/D (16x16 f32): VGPR r -> row M = r + (lane>=16)*8, col N = lane%16
__device__ __forceinline__ int a_kmap(int e, int lane) {
  return ((lane >> 4) << 4) + e;
}

// ---------------------------------------------------------------------------
// Packed-B index helper: Bp[((tile*nkb + kb)*32 + lane)*16 + e]
//   holds B element (k = kb*32 + (lane>=16)*16 + e, n = tile*16 + lane%16)
// ---------------------------------------------------------------------------
__device__ __forceinline__ void unpack_b_idx(long long idx, int nkb,
                                             int& tile, int& kb, int& k, int& n) {
  int e = (int)(idx & 15);
  int lane = (int)((idx >> 4) & 31);
  long long g = idx >> 9;
  kb = (int)(g % nkb);
  tile = (int)(g / nkb);
  k = kb * 32 + ((lane >> 4) << 4) + e;
  n = tile * 16 + (lane & 15);
}

// Conv2d weight [Cout,Cin,4,4] -> packed B, k = kh*4*Cin + kw*Cin + ci
__global__ void pack_conv_w(const float* __restrict__ w, _Float16* __restrict__ Bp,
                            int Cin, int Cout) {
  int Ktot = 16 * Cin;
  int nkb = ((Ktot + 31) & ~31) >> 5;
  long long tot = (long long)(Cout >> 4) * nkb * 512;
  long long idx = (long long)blockIdx.x * blockDim.x + threadIdx.x;
  if (idx >= tot) return;
  int tile, kb, k, n;
  unpack_b_idx(idx, nkb, tile, kb, k, n);
  float v = 0.0f;
  if (k < Ktot && n < Cout) {
    int q = k / Cin, ci = k - q * Cin;
    int kh = q >> 2, kw = q & 3;
    v = w[(((size_t)n * Cin + ci) * 4 + kh) * 4 + kw];
  }
  Bp[idx] = (_Float16)v;
}

// ConvTranspose2d weight [Cin,Cout,4,4] -> 4 parity classes, packed.
// class (ph,pw): k = (th*2+tw)*Cin + ci, kh = (1-ph)+2*th, kw = (1-pw)+2*tw
__global__ void pack_deconv_w(const float* __restrict__ w, _Float16* __restrict__ Bp,
                              int Cin, int Cout, int ntiles) {
  int Ktot = 4 * Cin;
  int nkb = ((Ktot + 31) & ~31) >> 5;
  long long per_cls = (long long)ntiles * nkb * 512;
  long long tot = 4 * per_cls;
  long long idx = (long long)blockIdx.x * blockDim.x + threadIdx.x;
  if (idx >= tot) return;
  int cls = (int)(idx / per_cls);
  int tile, kb, k, n;
  unpack_b_idx(idx - cls * per_cls, nkb, tile, kb, k, n);
  float v = 0.0f;
  if (k < Ktot && n < Cout) {
    int ph = cls >> 1, pw = cls & 1;
    int t = k / Cin, ci = k - t * Cin;
    int th = t >> 1, tw = t & 1;
    int kh = (1 - ph) + 2 * th, kw = (1 - pw) + 2 * tw;
    v = w[(((size_t)ci * Cout + n) * 4 + kh) * 4 + kw];
  }
  Bp[idx] = (_Float16)v;
}

// Generic strided gather, packed: B[k][n] = src[k*rs + n*cs]
__global__ void pack_mat(const float* __restrict__ src, _Float16* __restrict__ Bp,
                         int K, int N, int rs, int cs) {
  int nkb = ((K + 31) & ~31) >> 5;
  long long tot = (long long)(((N + 15) & ~15) >> 4) * nkb * 512;
  long long idx = (long long)blockIdx.x * blockDim.x + threadIdx.x;
  if (idx >= tot) return;
  int tile, kb, k, n;
  unpack_b_idx(idx, nkb, tile, kb, k, n);
  float v = (k < K && n < N) ? src[(size_t)k * rs + (size_t)n * cs] : 0.0f;
  Bp[idx] = (_Float16)v;
}

// conv1 im2col: x [128,1,64,64] -> P [131072, 32] fp32 (k<16 real, rest 0)
__global__ void im2col1(const float* __restrict__ x, float* __restrict__ P) {
  long long idx = (long long)blockIdx.x * blockDim.x + threadIdx.x;
  if (idx >= 131072LL * 32) return;
  int m = (int)(idx >> 5), k = (int)(idx & 31);
  float v = 0.0f;
  if (k < 16) {
    int b = m >> 10;                 // HO*WO = 32*32
    int r = m & 1023;
    int ho = r >> 5, wo = r & 31;
    int kh = k >> 2, kw = k & 3;
    int hi = 2 * ho - 1 + kh, wi = 2 * wo - 1 + kw;
    if ((unsigned)hi < 64u && (unsigned)wi < 64u)
      v = x[((size_t)b * 64 + hi) * 64 + wi];
  }
  P[idx] = v;
}

// ---------------------------------------------------------------------------
// 4-wave WMMA GEMM: C = act(A[M,lda] x Bpacked + bias)
// Requires Ktot % 32 == 0, lda % 4 == 0. Block: 16 rows x 256 cols.
// ---------------------------------------------------------------------------
__global__ void __launch_bounds__(128)
gemm4w(const float* __restrict__ A, int lda, int M, int Ktot,
       const _Float16* __restrict__ Bp, int ntilesTot,
       const float* __restrict__ bias, int N,
       float* __restrict__ C, int ldc, int act) {
  __shared__ __align__(32) _Float16 aSh[512];
  const int tid = threadIdx.x, lane = tid & 31, wave = tid >> 5;
  const int tM = blockIdx.y, tileBase = blockIdx.x * 16;
  const int nkb = Ktot >> 5;

  // staging: this thread supplies 4 consecutive halves h0..h0+3
  const int h0 = tid * 4;
  const int sLane = h0 >> 4, sE0 = h0 & 15;
  const int sMrow = tM * 16 + (sLane & 15);
  const bool sMval = sMrow < M;
  const int kOff = a_kmap(sE0, sLane);   // consecutive for e0..e0+3

  // unconditional B pointers (clamped tile; results masked at store time)
  const _Float16* bPtr[4];
#pragma unroll
  for (int j = 0; j < 4; ++j) {
    int tile = tileBase + wave * 4 + j;
    if (tile >= ntilesTot) tile = ntilesTot - 1;
    bPtr[j] = Bp + (size_t)tile * nkb * 512 + lane * 16;
  }

  v8f acc[4] = {{}, {}, {}, {}};
  for (int kb = 0; kb < nkb; ++kb) {
    __syncthreads();
    v4f v = {};
    if (sMval) v = *(const v4f*)(A + (size_t)sMrow * lda + kb * 32 + kOff);
    aSh[h0 + 0] = (_Float16)v.x; aSh[h0 + 1] = (_Float16)v.y;
    aSh[h0 + 2] = (_Float16)v.z; aSh[h0 + 3] = (_Float16)v.w;
    __syncthreads();
    v16h a = *(const v16h*)&aSh[lane * 16];
#pragma unroll
    for (int j = 0; j < 4; ++j) {
      v16h b = *(const v16h*)(bPtr[j] + (size_t)kb * 512);
      acc[j] = __builtin_amdgcn_wmma_f32_16x16x32_f16(false, a, false, b,
                                                      (short)0, acc[j], false, false);
    }
  }
#pragma unroll
  for (int j = 0; j < 4; ++j) {
    int tile = tileBase + wave * 4 + j;
    if (tile >= ntilesTot) continue;
    int col = tile * 16 + (lane & 15);
    if (col >= N) continue;
    float bv = bias ? bias[col] : 0.0f;
#pragma unroll
    for (int r = 0; r < 8; ++r) {
      int row = tM * 16 + ((lane >> 4) << 3) + r;
      if (row < M) {
        float o = acc[j][r] + bv;
        if (act == 1) o = fmaxf(o, 0.0f);
        C[(size_t)row * ldc + col] = o;
      }
    }
  }
}

// ---------------------------------------------------------------------------
// 4-wave implicit-GEMM Conv2d(k=4,s=2,p=1), NHWC, Cin = power of 2, %32 == 0.
// Each 32-wide K block sits inside one (kh,kw) tap -> contiguous channels.
// ---------------------------------------------------------------------------
__global__ void __launch_bounds__(128)
conv4w(const float* __restrict__ in, const _Float16* __restrict__ Bp,
       const float* __restrict__ bias, float* __restrict__ out,
       int Bn, int H, int W, int Cin, int cinShift,
       int HO, int WO, int Cout, int act) {
  __shared__ __align__(32) _Float16 aSh[512];
  const int tid = threadIdx.x, lane = tid & 31, wave = tid >> 5;
  const int tM = blockIdx.y;
  const int M = Bn * HO * WO;
  const int nkb = (16 * Cin) >> 5;
  const int ntilesTot = Cout >> 4;

  const int h0 = tid * 4;
  const int sLane = h0 >> 4, sE0 = h0 & 15;
  const int sMrow = tM * 16 + (sLane & 15);
  const bool sMval = sMrow < M;
  int b_ = 0, ho = 0, wo = 0;
  if (sMval) {
    b_ = sMrow / (HO * WO);
    int r = sMrow - b_ * (HO * WO);
    ho = r / WO; wo = r - ho * WO;
  }
  const int kOff = a_kmap(sE0, sLane);

  const _Float16* bPtr[4];
#pragma unroll
  for (int j = 0; j < 4; ++j) {
    int tile = wave * 4 + j;
    if (tile >= ntilesTot) tile = ntilesTot - 1;
    bPtr[j] = Bp + (size_t)tile * nkb * 512 + lane * 16;
  }

  v8f acc[4] = {{}, {}, {}, {}};
  for (int kb = 0; kb < nkb; ++kb) {
    __syncthreads();
    int kk = kb * 32 + kOff;
    int q = kk >> cinShift;            // tap 0..15 (uniform over the 4 halves)
    int ci = kk & (Cin - 1);
    int kh = q >> 2, kw = q & 3;
    int hi = 2 * ho - 1 + kh, wi = 2 * wo - 1 + kw;
    v4f v = {};
    if (sMval && (unsigned)hi < (unsigned)H && (unsigned)wi < (unsigned)W)
      v = *(const v4f*)(in + (((size_t)b_ * H + hi) * W + wi) * Cin + ci);
    aSh[h0 + 0] = (_Float16)v.x; aSh[h0 + 1] = (_Float16)v.y;
    aSh[h0 + 2] = (_Float16)v.z; aSh[h0 + 3] = (_Float16)v.w;
    __syncthreads();
    v16h a = *(const v16h*)&aSh[lane * 16];
#pragma unroll
    for (int j = 0; j < 4; ++j) {
      v16h b = *(const v16h*)(bPtr[j] + (size_t)kb * 512);
      acc[j] = __builtin_amdgcn_wmma_f32_16x16x32_f16(false, a, false, b,
                                                      (short)0, acc[j], false, false);
    }
  }
#pragma unroll
  for (int j = 0; j < 4; ++j) {
    int tile = wave * 4 + j;
    if (tile >= ntilesTot) continue;
    int col = tile * 16 + (lane & 15);
    if (col >= Cout) continue;
    float bv = bias[col];
#pragma unroll
    for (int r = 0; r < 8; ++r) {
      int row = tM * 16 + ((lane >> 4) << 3) + r;
      if (row < M) {
        float o = acc[j][r] + bv;
        if (act == 1) o = fmaxf(o, 0.0f);
        out[(size_t)row * Cout + col] = o;
      }
    }
  }
}

// ---------------------------------------------------------------------------
// 4-wave implicit-GEMM ConvTranspose2d(k=4,s=2,p=1), 4 parity classes (grid z).
// class (ph,pw): out(2q+p) gathers in(q+p-th) vs subkernel kh=(1-ph)+2th.
// ---------------------------------------------------------------------------
__global__ void __launch_bounds__(128)
deconv4w(const float* __restrict__ in, const _Float16* __restrict__ Bp,
         const float* __restrict__ bias, float* __restrict__ out,
         int Bn, int HI, int WI, int Cin, int cinShift,
         int Cout, int ntilesTot, int act) {
  __shared__ __align__(32) _Float16 aSh[512];
  const int tid = threadIdx.x, lane = tid & 31, wave = tid >> 5;
  const int tM = blockIdx.y;
  const int cls = blockIdx.z;
  const int ph = cls >> 1, pw = cls & 1;
  const int M = Bn * HI * WI;
  const int nkb = (4 * Cin) >> 5;
  const _Float16* Bc = Bp + (size_t)cls * ntilesTot * nkb * 512;

  const int h0 = tid * 4;
  const int sLane = h0 >> 4, sE0 = h0 & 15;
  const int sMrow = tM * 16 + (sLane & 15);
  const bool sMval = sMrow < M;
  int b_ = 0, qh = 0, qw = 0;
  if (sMval) {
    b_ = sMrow / (HI * WI);
    int r = sMrow - b_ * (HI * WI);
    qh = r / WI; qw = r - qh * WI;
  }
  const int kOff = a_kmap(sE0, sLane);

  const _Float16* bPtr[4];
#pragma unroll
  for (int j = 0; j < 4; ++j) {
    int tile = wave * 4 + j;
    if (tile >= ntilesTot) tile = ntilesTot - 1;
    bPtr[j] = Bc + (size_t)tile * nkb * 512 + lane * 16;
  }

  v8f acc[4] = {{}, {}, {}, {}};
  for (int kb = 0; kb < nkb; ++kb) {
    __syncthreads();
    int kk = kb * 32 + kOff;
    int t = kk >> cinShift;            // tap 0..3
    int ci = kk & (Cin - 1);
    int th = t >> 1, tw = t & 1;
    int hi = qh + ph - th, wi = qw + pw - tw;
    v4f v = {};
    if (sMval && (unsigned)hi < (unsigned)HI && (unsigned)wi < (unsigned)WI)
      v = *(const v4f*)(in + (((size_t)b_ * HI + hi) * WI + wi) * Cin + ci);
    aSh[h0 + 0] = (_Float16)v.x; aSh[h0 + 1] = (_Float16)v.y;
    aSh[h0 + 2] = (_Float16)v.z; aSh[h0 + 3] = (_Float16)v.w;
    __syncthreads();
    v16h a = *(const v16h*)&aSh[lane * 16];
#pragma unroll
    for (int j = 0; j < 4; ++j) {
      v16h b = *(const v16h*)(bPtr[j] + (size_t)kb * 512);
      acc[j] = __builtin_amdgcn_wmma_f32_16x16x32_f16(false, a, false, b,
                                                      (short)0, acc[j], false, false);
    }
  }
  const int HO = 2 * HI, WO = 2 * WI;
#pragma unroll
  for (int j = 0; j < 4; ++j) {
    int tile = wave * 4 + j;
    if (tile >= ntilesTot) continue;
    int col = tile * 16 + (lane & 15);
    if (col >= Cout) continue;
    float bv = bias[col];
#pragma unroll
    for (int r = 0; r < 8; ++r) {
      int row = tM * 16 + ((lane >> 4) << 3) + r;
      if (row < M) {
        int bb = row / (HI * WI);
        int rr = row - bb * (HI * WI);
        int h = rr / WI, w = rr - h * WI;
        float o = acc[j][r] + bv;
        if (act == 1) o = fmaxf(o, 0.0f);
        else if (act == 2) o = 1.0f / (1.0f + expf(-o));
        out[(((size_t)bb * HO + 2 * h + ph) * WO + 2 * w + pw) * Cout + col] = o;
      }
    }
  }
}

// ---------------------------------------------------------------------------
// VQ / misc helpers (deterministic reductions)
// ---------------------------------------------------------------------------
__global__ void cb_sq_k(const float* __restrict__ cb, float* __restrict__ cbsq) {
  int n = blockIdx.x * blockDim.x + threadIdx.x;
  if (n >= 4096) return;
  float s = 0.0f;
  for (int d = 0; d < 256; ++d) { float v = cb[(size_t)n * 256 + d]; s += v * v; }
  cbsq[n] = s;
}

__global__ void vq_argmin_k(const float* __restrict__ G, const float* __restrict__ cbsq,
                            int* __restrict__ codes, int NCc) {
  int m = blockIdx.x, t = threadIdx.x;
  float best = 3.4e38f; int bi = NCc;
  for (int n = t; n < NCc; n += blockDim.x) {
    float d = cbsq[n] - 2.0f * G[(size_t)m * NCc + n];
    if (d < best) { best = d; bi = n; }
  }
  __shared__ float sv[256];
  __shared__ int si[256];
  sv[t] = best; si[t] = bi;
  __syncthreads();
  for (int s = 128; s > 0; s >>= 1) {
    if (t < s) {
      if (sv[t + s] < sv[t] || (sv[t + s] == sv[t] && si[t + s] < si[t])) {
        sv[t] = sv[t + s]; si[t] = si[t + s];
      }
    }
    __syncthreads();
  }
  if (t == 0) codes[m] = si[0];
}

__global__ void codes_to_f(const int* __restrict__ codes, float* __restrict__ dst, int n) {
  int i = blockIdx.x * blockDim.x + threadIdx.x;
  if (i < n) dst[i] = (float)codes[i];
}

__global__ void gather_zq_loss(const float* __restrict__ zf, const float* __restrict__ cb,
                               const int* __restrict__ codes, float* __restrict__ zq,
                               float* __restrict__ partial) {
  int m = blockIdx.x, d = threadIdx.x;  // 256 threads
  float a = zf[(size_t)m * 256 + d];
  float q = cb[(size_t)codes[m] * 256 + d];
  zq[(size_t)m * 256 + d] = q;
  float diff = a - q;
  __shared__ float s[256];
  s[d] = diff * diff;
  __syncthreads();
  for (int st = 128; st > 0; st >>= 1) {
    if (d < st) s[d] += s[d + st];
    __syncthreads();
  }
  if (d == 0) partial[m] = s[0];
}

__global__ void finalize_loss(const float* __restrict__ partial, float* __restrict__ out3) {
  int t = threadIdx.x;
  __shared__ float s[256];
  float acc = 0.0f;
  for (int i = t; i < 2048; i += 256) acc += partial[i];
  s[t] = acc;
  __syncthreads();
  for (int st = 128; st > 0; st >>= 1) {
    if (t < st) s[t] += s[t + st];
    __syncthreads();
  }
  if (t == 0) {
    float mse = s[0] / (2048.0f * 256.0f);
    out3[0] = mse;          // commitment_loss (== codebook_loss numerically fwd)
    out3[1] = mse;          // codebook_loss
    out3[2] = 1.25f * mse;  // vq = codebook + 0.25*commitment
  }
}

__global__ void build_dyn(const int* __restrict__ codes, const float* __restrict__ code_embed,
                          const float* __restrict__ action_embed, const int* __restrict__ action,
                          float* __restrict__ hcat) {
  int b = blockIdx.x;
  for (int j = threadIdx.x; j < 2176; j += blockDim.x) {
    float v;
    if (j < 2048) {
      int p = j >> 7, e = j & 127;
      v = code_embed[(size_t)codes[b * 16 + p] * 128 + e];
    } else {
      v = action_embed[(size_t)action[b] * 128 + (j - 2048)];
    }
    hcat[(size_t)b * 2176 + j] = v;
  }
}

// ---------------------------------------------------------------------------
// Orchestration
// ---------------------------------------------------------------------------
extern "C" void kernel_launch(void* const* d_in, const int* in_sizes, int n_in,
                              void* d_out, int out_size, void* d_ws, size_t ws_size,
                              hipStream_t stream) {
  (void)in_sizes; (void)n_in; (void)out_size; (void)ws_size;

  const float* x          = (const float*)d_in[0];
  const float* x_next     = (const float*)d_in[1];
  const int*   action     = (const int*)  d_in[2];
  const float* enc_w1 = (const float*)d_in[3];  const float* enc_b1 = (const float*)d_in[4];
  const float* enc_w2 = (const float*)d_in[5];  const float* enc_b2 = (const float*)d_in[6];
  const float* enc_w3 = (const float*)d_in[7];  const float* enc_b3 = (const float*)d_in[8];
  const float* enc_w4 = (const float*)d_in[9];  const float* enc_b4 = (const float*)d_in[10];
  const float* dec_w1 = (const float*)d_in[11]; const float* dec_b1 = (const float*)d_in[12];
  const float* dec_w2 = (const float*)d_in[13]; const float* dec_b2 = (const float*)d_in[14];
  const float* dec_w3 = (const float*)d_in[15]; const float* dec_b3 = (const float*)d_in[16];
  const float* dec_w4 = (const float*)d_in[17]; const float* dec_b4 = (const float*)d_in[18];
  const float* codebook     = (const float*)d_in[19];
  const float* code_embed   = (const float*)d_in[20];
  const float* action_embed = (const float*)d_in[21];
  const float* dyn_w1 = (const float*)d_in[22]; const float* dyn_b1 = (const float*)d_in[23];
  const float* dyn_w2 = (const float*)d_in[24]; const float* dyn_b2 = (const float*)d_in[25];
  const float* dyn_w3 = (const float*)d_in[26]; const float* dyn_b3 = (const float*)d_in[27];

  float* out = (float*)d_out;
  const size_t OFF_RECON = 0;
  const size_t OFF_CODES = 524288;      // 128*1*64*64
  const size_t OFF_CLOSS = 526336;      // + 128*4*4
  const size_t OFF_NLOG  = 526339;      // + 3 scalars
  const size_t OFF_NCT   = 8914947;     // + 128*4*4*4096

  char* base = (char*)d_ws;
  size_t off = 0;
  auto alloc = [&](size_t bytes) -> void* {
    void* p = base + off;
    off += (bytes + 255) & ~(size_t)255;
    return p;
  };
  float* e1 = (float*)alloc(33554432ull * 4);   // [128,32,32,256] NHWC (alias d3)
  float* e2 = (float*)alloc(8388608ull * 4);    // [128,16,16,256]      (alias d2)
  float* e3 = (float*)alloc(2097152ull * 4);    // [128,8,8,256]        (alias d1)
  float* z  = (float*)alloc(524288ull * 4);     // zf [2048,256]
  float* G  = (float*)alloc(8388608ull * 4);    // zf @ cb^T [2048,4096]
  float* zq = (float*)alloc(524288ull * 4);
  float* P1 = (float*)alloc(131072ull * 32 * 4);// conv1 im2col patches
  float* hcat = (float*)alloc(128ull * 2176 * 4);
  float* h1 = (float*)alloc(32768ull * 4);
  float* h2 = (float*)alloc(32768ull * 4);
  float* cbsq = (float*)alloc(4096ull * 4);
  float* partial = (float*)alloc(2048ull * 4);
  int* codesA = (int*)alloc(2048ull * 4);
  int* codesB = (int*)alloc(2048ull * 4);
  _Float16* enc1p = (_Float16*)alloc(16ull * 1 * 512 * 2);
  _Float16* enc2p = (_Float16*)alloc(16ull * 128 * 512 * 2);
  _Float16* enc3p = (_Float16*)alloc(16ull * 128 * 512 * 2);
  _Float16* enc4p = (_Float16*)alloc(16ull * 128 * 512 * 2);
  _Float16* cbTp  = (_Float16*)alloc(256ull * 8 * 512 * 2);
  _Float16* dec1p = (_Float16*)alloc(4ull * 16 * 32 * 512 * 2);
  _Float16* dec2p = (_Float16*)alloc(4ull * 16 * 32 * 512 * 2);
  _Float16* dec3p = (_Float16*)alloc(4ull * 16 * 32 * 512 * 2);
  _Float16* dec4p = (_Float16*)alloc(4ull * 1 * 32 * 512 * 2);
  _Float16* dynW1p = (_Float16*)alloc(16ull * 68 * 512 * 2);
  _Float16* dynW2p = (_Float16*)alloc(16ull * 8 * 512 * 2);
  _Float16* dynW3p = (_Float16*)alloc(4096ull * 8 * 512 * 2);
  float* d1 = e3;  // decoder aliases encoder scratch (encoder done first)
  float* d2 = e2;
  float* d3 = e1;

  // --- weight packing (fp32 -> fp16 WMMA-fragment order) ---
  pack_conv_w<<<dim3(32), 256, 0, stream>>>(enc_w1, enc1p, 1, 256);
  pack_conv_w<<<dim3(4096), 256, 0, stream>>>(enc_w2, enc2p, 256, 256);
  pack_conv_w<<<dim3(4096), 256, 0, stream>>>(enc_w3, enc3p, 256, 256);
  pack_conv_w<<<dim3(4096), 256, 0, stream>>>(enc_w4, enc4p, 256, 256);
  pack_deconv_w<<<dim3(4096), 256, 0, stream>>>(dec_w1, dec1p, 256, 256, 16);
  pack_deconv_w<<<dim3(4096), 256, 0, stream>>>(dec_w2, dec2p, 256, 256, 16);
  pack_deconv_w<<<dim3(4096), 256, 0, stream>>>(dec_w3, dec3p, 256, 256, 16);
  pack_deconv_w<<<dim3(256), 256, 0, stream>>>(dec_w4, dec4p, 256, 1, 1);
  pack_mat<<<dim3(4096), 256, 0, stream>>>(codebook, cbTp, 256, 4096, 1, 256);
  pack_mat<<<dim3(2176), 256, 0, stream>>>(dyn_w1, dynW1p, 2176, 256, 256, 1);
  pack_mat<<<dim3(256), 256, 0, stream>>>(dyn_w2, dynW2p, 256, 256, 256, 1);
  pack_mat<<<dim3(65536), 256, 0, stream>>>(dyn_w3, dynW3p, 256, 65536, 65536, 1);
  cb_sq_k<<<dim3(16), 256, 0, stream>>>(codebook, cbsq);

  auto encode = [&](const float* xin, int* codes) {
    im2col1<<<dim3(16384), 256, 0, stream>>>(xin, P1);
    gemm4w<<<dim3(1, 8192), 128, 0, stream>>>(P1, 32, 131072, 32, enc1p, 16,
                                              enc_b1, 256, e1, 256, 1);
    conv4w<<<dim3(1, 2048), 128, 0, stream>>>(e1, enc2p, enc_b2, e2,
                                              128, 32, 32, 256, 8, 16, 16, 256, 1);
    conv4w<<<dim3(1, 512), 128, 0, stream>>>(e2, enc3p, enc_b3, e3,
                                             128, 16, 16, 256, 8, 8, 8, 256, 1);
    conv4w<<<dim3(1, 128), 128, 0, stream>>>(e3, enc4p, enc_b4, z,
                                             128, 8, 8, 256, 8, 4, 4, 256, 0);
    gemm4w<<<dim3(16, 128), 128, 0, stream>>>(z, 256, 2048, 256, cbTp, 256,
                                              nullptr, 4096, G, 4096, 0);
    vq_argmin_k<<<dim3(2048), 256, 0, stream>>>(G, cbsq, codes, 4096);
  };

  // --- encode x, VQ, losses ---
  encode(x, codesA);
  codes_to_f<<<dim3(8), 256, 0, stream>>>(codesA, out + OFF_CODES, 2048);
  gather_zq_loss<<<dim3(2048), 256, 0, stream>>>(z, codebook, codesA, zq, partial);
  finalize_loss<<<dim3(1), 256, 0, stream>>>(partial, out + OFF_CLOSS);

  // --- decoder (zq_st == zq numerically; straight-through is grad-only) ---
  deconv4w<<<dim3(1, 128, 4), 128, 0, stream>>>(zq, dec1p, dec_b1, d1,
                                                128, 4, 4, 256, 8, 256, 16, 1);
  deconv4w<<<dim3(1, 512, 4), 128, 0, stream>>>(d1, dec2p, dec_b2, d2,
                                                128, 8, 8, 256, 8, 256, 16, 1);
  deconv4w<<<dim3(1, 2048, 4), 128, 0, stream>>>(d2, dec3p, dec_b3, d3,
                                                 128, 16, 16, 256, 8, 256, 16, 1);
  deconv4w<<<dim3(1, 8192, 4), 128, 0, stream>>>(d3, dec4p, dec_b4, out + OFF_RECON,
                                                 128, 32, 32, 256, 8, 1, 1, 2);

  // --- dynamics MLP ---
  build_dyn<<<dim3(128), 256, 0, stream>>>(codesA, code_embed, action_embed, action, hcat);
  gemm4w<<<dim3(1, 8), 128, 0, stream>>>(hcat, 2176, 128, 2176, dynW1p, 16,
                                         dyn_b1, 256, h1, 256, 1);
  gemm4w<<<dim3(1, 8), 128, 0, stream>>>(h1, 256, 128, 256, dynW2p, 16,
                                         dyn_b2, 256, h2, 256, 1);
  gemm4w<<<dim3(256, 8), 128, 0, stream>>>(h2, 256, 128, 256, dynW3p, 4096,
                                           dyn_b3, 65536, out + OFF_NLOG, 65536, 0);

  // --- encode x_next for next-code targets (reuses encoder scratch) ---
  encode(x_next, codesB);
  codes_to_f<<<dim3(8), 256, 0, stream>>>(codesB, out + OFF_NCT, 2048);
}
